// MC_Loss_67190468379033
// MI455X (gfx1250) — compile-verified, matrix operand order checked by reading, not run on hardware
//
#include <hip/hip_runtime.h>
#include <hip/hip_bf16.h>

typedef __attribute__((ext_vector_type(16))) __bf16 v16bf;
typedef __attribute__((ext_vector_type(8)))  __bf16 v8bf;
typedef __attribute__((ext_vector_type(4)))  __bf16 v4bf;
typedef __attribute__((ext_vector_type(8)))  float  v8f;

union bfrag { v16bf v; v8bf h[2]; };

// ---------------------------------------------------------------------------
// 1) fp32 -> bf16 feature conversion (src|tgt|gen concatenated in ws)
// ---------------------------------------------------------------------------
__global__ void convert_bf16_kernel(const float* __restrict__ s,
                                    const float* __restrict__ t,
                                    const float* __restrict__ g,
                                    __bf16* __restrict__ out) {
  const int idx = blockIdx.x * 256 + threadIdx.x;      // 0 .. 3*2^20-1
  const int region = idx >> 20;
  const int w = idx & 1048575;
  const float* p = (region == 0) ? s : ((region == 1) ? t : g);
  out[idx] = (__bf16)p[w];
}

// ---------------------------------------------------------------------------
// 2) Cost GEMM + kernelization: K[g,i,o] = exp(1 - A[g,i,:]·B[g,o,:]),
//    diagonal = exp(-10).  g in 0..7: g<4 -> (src,tgt) pair b=g (for f1),
//    g>=4 -> (tgt,gen) pair b=g-4 (for f2).  One wave per 16x16 tile,
//    4x v_wmma_f32_16x16x32_bf16 over d=128.
// ---------------------------------------------------------------------------
__global__ void cost_gemm_kernel(const __bf16* __restrict__ srcb,
                                 const __bf16* __restrict__ tgtb,
                                 const __bf16* __restrict__ genb,
                                 __bf16* __restrict__ K) {
  const int lane = threadIdx.x & 31;
  const int wave = threadIdx.x >> 5;
  const int tilesPerG = 128 * 128;
  int t = blockIdx.x * 8 + wave;               // global tile id, 0..131071
  const int g = t / tilesPerG;
  t -= g * tilesPerG;
  const int row0 = (t >> 7) << 4;
  const int col0 = (t & 127) << 4;
  const int half = lane >> 4;
  const int r    = lane & 15;

  const __bf16* A = (g < 4) ? (srcb + (size_t)g       * 2048 * 128)
                            : (tgtb + (size_t)(g - 4) * 2048 * 128);
  const __bf16* B = (g < 4) ? (tgtb + (size_t)g       * 2048 * 128)
                            : (genb + (size_t)(g - 4) * 2048 * 128);

  const __bf16* arow = A + (size_t)(row0 + r) * 128;
  const __bf16* brow = B + (size_t)(col0 + r) * 128;

  v8f c = {};
#pragma unroll
  for (int kc = 0; kc < 4; ++kc) {
    const int kb = kc * 32;
    bfrag a, b;
    // 16-bit A/B fragment layout (ISA 7.12.2): lane<16 holds K=kb+[0..7],
    // K=kb+16+[0..7]; lane>=16 holds K=kb+8+[0..7], K=kb+24+[0..7].
    a.h[0] = *(const v8bf*)(arow + kb +      half * 8);
    a.h[1] = *(const v8bf*)(arow + kb + 16 + half * 8);
    b.h[0] = *(const v8bf*)(brow + kb +      half * 8);
    b.h[1] = *(const v8bf*)(brow + kb + 16 + half * 8);
    c = __builtin_amdgcn_wmma_f32_16x16x32_bf16(
        /*neg_a=*/false, a.v, /*neg_b=*/false, b.v,
        /*c_mod=*/(short)0, c, /*reuse_a=*/false, /*reuse_b=*/false);
  }

  __bf16* Kg = K + ((size_t)g << 22);
#pragma unroll
  for (int j = 0; j < 8; ++j) {
    const int m = row0 + j + 8 * half;   // C/D layout: VGPR j -> M=j / M=8+j
    const int n = col0 + r;              // N = lane & 15
    const float val = (m == n) ? __expf(-10.0f) : __expf(1.0f - c[j]);
    Kg[((size_t)m << 11) + n] = (__bf16)val;
  }
}

// ---------------------------------------------------------------------------
// 3) Sinkhorn
// ---------------------------------------------------------------------------
__global__ void init_v_kernel(float* __restrict__ v) {
  v[blockIdx.x * 256 + threadIdx.x] = 1.0f;
}

// u[g,i] = 1 / sum_o K[g,i,o] * v[g,o]   (one wave per row, row-major sweep)
__global__ void u_step_kernel(const __bf16* __restrict__ K,
                              const float* __restrict__ v,
                              float* __restrict__ u) {
  const int lane = threadIdx.x & 31;
  const int wave = threadIdx.x >> 5;
  const int gi = blockIdx.x * 8 + wave;        // 0..16383
  const int g = gi >> 11;
  const __bf16* Krow = K + ((size_t)gi << 11);
  const float* vg = v + (g << 11);
  float acc = 0.f;
#pragma unroll
  for (int s = 0; s < 8; ++s) {
    const int idx = s * 256 + lane * 8;
    v8bf k8 = *(const v8bf*)(Krow + idx);
    const float4 f0 = *(const float4*)(vg + idx);
    const float4 f1 = *(const float4*)(vg + idx + 4);
    acc += (float)k8[0] * f0.x + (float)k8[1] * f0.y +
           (float)k8[2] * f0.z + (float)k8[3] * f0.w +
           (float)k8[4] * f1.x + (float)k8[5] * f1.y +
           (float)k8[6] * f1.z + (float)k8[7] * f1.w;
  }
#pragma unroll
  for (int off = 16; off >= 1; off >>= 1) acc += __shfl_xor(acc, off, 32);
  if (lane == 0) u[gi] = 1.0f / acc;           // a = out/in = 1
}

// partial[g,rc,o] = sum_{i in chunk rc} u[g,i] * K[g,i,o]
// 16 chunks of 128 rows; block = 256 threads x 4 cols (1024 cols), cb in {0,1}
__global__ void v_partial_kernel(const __bf16* __restrict__ K,
                                 const float* __restrict__ u,
                                 float* __restrict__ part) {
  __shared__ float su[128];
  const int bid = blockIdx.x;                  // 0..255
  const int g  = bid >> 5;
  const int rc = (bid >> 1) & 15;
  const int cb = bid & 1;
  const int tid = threadIdx.x;
  if (tid < 128) su[tid] = u[(g << 11) + rc * 128 + tid];
  __syncthreads();
  const int col = cb * 1024 + tid * 4;
  const __bf16* Kp = K + ((size_t)g << 22) + ((size_t)(rc * 128) << 11) + col;
  float a0 = 0.f, a1 = 0.f, a2 = 0.f, a3 = 0.f;
  for (int ii = 0; ii < 128; ++ii) {
    v4bf k4 = *(const v4bf*)(Kp + ((size_t)ii << 11));
    const float uu = su[ii];
    a0 += uu * (float)k4[0];
    a1 += uu * (float)k4[1];
    a2 += uu * (float)k4[2];
    a3 += uu * (float)k4[3];
  }
  float* p = part + ((size_t)(g * 16 + rc) << 11) + col;
  p[0] = a0; p[1] = a1; p[2] = a2; p[3] = a3;
}

// v[g,o] = 1 / sum_rc partial[g,rc,o]   (fixed-order -> deterministic)
__global__ void v_reduce_kernel(const float* __restrict__ part,
                                float* __restrict__ v) {
  const int idx = blockIdx.x * 256 + threadIdx.x;   // 0..16383
  const int g = idx >> 11, o = idx & 2047;
  float s = 0.f;
#pragma unroll
  for (int rc = 0; rc < 16; ++rc) s += part[((size_t)(g * 16 + rc) << 11) + o];
  v[idx] = 1.0f / s;
}

// ---------------------------------------------------------------------------
// 4) loss = mean | u1*K1*v1 - u2*K2*v2 |  (transpose cancels elementwise)
// ---------------------------------------------------------------------------
__global__ void loss_partial_kernel(const __bf16* __restrict__ K,
                                    const float* __restrict__ u,
                                    const float* __restrict__ v,
                                    float* __restrict__ part) {
  __shared__ float sdata[256];
  const int tid = threadIdx.x;
  float acc = 0.f;
  const size_t nchunk = 2097152;               // (4*2048*2048)/8
  for (size_t c = (size_t)blockIdx.x * 256 + tid; c < nchunk;
       c += (size_t)gridDim.x * 256) {
    const size_t e = c * 8;
    const int b = (int)(e >> 22);
    const size_t rem = e & ((size_t)(1u << 22) - 1);
    const int i = (int)(rem >> 11);
    const int o = (int)(rem & 2047);
    v8bf x1 = *(const v8bf*)(K + ((size_t)b << 22) + rem);
    v8bf x2 = *(const v8bf*)(K + ((size_t)(b + 4) << 22) + rem);
    const float u1 = u[(b << 11) + i];
    const float u2 = u[((b + 4) << 11) + i];
    const float* v1 = v + (b << 11) + o;
    const float* v2 = v + ((b + 4) << 11) + o;
#pragma unroll
    for (int j = 0; j < 8; ++j)
      acc += fabsf(u1 * (float)x1[j] * v1[j] - u2 * (float)x2[j] * v2[j]);
  }
  sdata[tid] = acc;
  __syncthreads();
  for (int s = 128; s > 0; s >>= 1) {
    if (tid < s) sdata[tid] += sdata[tid + s];
    __syncthreads();
  }
  if (tid == 0) part[blockIdx.x] = sdata[0];
}

__global__ void loss_final_kernel(const float* __restrict__ part,
                                  float* __restrict__ out) {
  __shared__ float sdata[256];
  const int tid = threadIdx.x;
  float acc = 0.f;
  for (int i = tid; i < 2048; i += 256) acc += part[i];
  sdata[tid] = acc;
  __syncthreads();
  for (int s = 128; s > 0; s >>= 1) {
    if (tid < s) sdata[tid] += sdata[tid + s];
    __syncthreads();
  }
  if (tid == 0) out[0] = sdata[0] / 16777216.0f;   // mean over 4*2048*2048
}

// ---------------------------------------------------------------------------
extern "C" void kernel_launch(void* const* d_in, const int* in_sizes, int n_in,
                              void* d_out, int out_size, void* d_ws, size_t ws_size,
                              hipStream_t stream) {
  const float* src = (const float*)d_in[0];
  const float* tgt = (const float*)d_in[1];
  const float* gen = (const float*)d_in[2];
  float* out = (float*)d_out;

  char* ws = (char*)d_ws;
  __bf16* K      = (__bf16*)(ws);                 // 8*2048*2048 bf16 = 64 MB
  __bf16* featbf = (__bf16*)(ws + 67108864);      // 3*2^20 bf16 = 6 MB
  float*  u      = (float*)(ws + 73400320);       // 16384 f32
  float*  v      = (float*)(ws + 73465856);       // 16384 f32
  float*  vpart  = (float*)(ws + 73531392);       // 8*16*2048 f32 = 1 MB
  float*  lpart  = (float*)(ws + 74579968);       // 2048 f32

  __bf16* srcb = featbf;
  __bf16* tgtb = featbf + 1048576;
  __bf16* genb = featbf + 2097152;

  // Features -> bf16
  convert_bf16_kernel<<<12288, 256, 0, stream>>>(src, tgt, gen, featbf);

  // K = exp(1 - q k^T), diag exp(-10), both OT problems (g=0..7)
  cost_gemm_kernel<<<16384, 256, 0, stream>>>(srcb, tgtb, genb, K);

  // Sinkhorn: v0 = 1; 50x { u = 1/(K v); v = 1/(u K) }
  init_v_kernel<<<64, 256, 0, stream>>>(v);
  for (int it = 0; it < 50; ++it) {
    u_step_kernel<<<2048, 256, 0, stream>>>(K, v, u);
    v_partial_kernel<<<256, 256, 0, stream>>>(K, u, vpart);
    v_reduce_kernel<<<64, 256, 0, stream>>>(vpart, v);
  }

  // loss = mean |u1 K1 v1 - u2 K2 v2|
  loss_partial_kernel<<<2048, 256, 0, stream>>>(K, u, v, lpart);
  loss_final_kernel<<<1, 256, 0, stream>>>(lpart, out);
}
